// EdgeDecoder_23751169146902
// MI455X (gfx1250) — matrix operand order repaired and checked by reading, not compiled, not run on hardware
//
#include <hip/hip_runtime.h>
#include <stdint.h>

// CDNA5 WMMA fragment types (probe-verified signatures)
typedef __attribute__((ext_vector_type(16))) __bf16 v16bf;
typedef __attribute__((ext_vector_type(8)))  float  v8f;

#define HDIM 128

union FragBF {
  v16bf    bf;
  uint32_t u[8];
  uint4    q[2];
};

// f32 x2 -> packed bf16 pair.
#if __has_builtin(__builtin_amdgcn_cvt_pk_bf16_f32)
// single v_cvt_pk_bf16_f32, round-to-nearest-even
__device__ __forceinline__ uint32_t pack2_bf16(float a, float b) {
  auto r = __builtin_amdgcn_cvt_pk_bf16_f32(a, b);
  uint32_t u;
  __builtin_memcpy(&u, &r, 4);
  return u;
}
#else
// fallback: round-half-up adds + one v_perm_b32 byte select
__device__ __forceinline__ uint32_t pack2_bf16(float a, float b) {
  uint32_t ua = __builtin_bit_cast(uint32_t, a) + 0x8000u;
  uint32_t ub = __builtin_bit_cast(uint32_t, b) + 0x8000u;
  return __builtin_amdgcn_perm(ub, ua, 0x07060302u);
}
#endif

// Build one 16x32 bf16 A-fragment for this lane (ISA 7.12.2, 16-bit A 16x32):
// lane L: M = L%16, half h = L/16; VGPR v: K = 16*(v/4) + 8*h + 2*(v%4) (+1 hi).
// rowp points at this edge's 32-float K-chunk.
__device__ __forceinline__ void gather_frag(FragBF& fr, const float* rowp, int h) {
  const float4* p = (const float4*)rowp;
  float4 q0 = p[2 * h + 0];      // K 8h   .. 8h+3
  float4 q1 = p[2 * h + 1];      // K 8h+4 .. 8h+7
  float4 q2 = p[4 + 2 * h + 0];  // K 16+8h .. +3
  float4 q3 = p[4 + 2 * h + 1];  // K 16+8h+4 .. +7
  fr.u[0] = pack2_bf16(q0.x, q0.y);
  fr.u[1] = pack2_bf16(q0.z, q0.w);
  fr.u[2] = pack2_bf16(q1.x, q1.y);
  fr.u[3] = pack2_bf16(q1.z, q1.w);
  fr.u[4] = pack2_bf16(q2.x, q2.y);
  fr.u[5] = pack2_bf16(q2.z, q2.w);
  fr.u[6] = pack2_bf16(q3.x, q3.y);
  fr.u[7] = pack2_bf16(q3.z, q3.w);
}

__global__ __launch_bounds__(128)
void edge_decoder_wmma(const float* __restrict__ zu,
                       const float* __restrict__ zm,
                       const int*   __restrict__ eidx,
                       const float* __restrict__ W1,
                       const float* __restrict__ b1,
                       const float* __restrict__ W2,
                       const float* __restrict__ b2,
                       float* __restrict__ out,
                       int E) {
  // W1^T, packed bf16 K-pairs. Row f holds 128 pairs, rotated by 4*(f&15) dwords
  // (keeps 4-dword groups contiguous -> ds_load_b128, bank-optimal). Exactly 64 KB.
  __shared__ uint32_t sW[HDIM * 128];

  const int tid  = threadIdx.x;   // 0..127 (4 waves)
  const int lane = tid & 31;
  const int wv   = tid >> 5;
  const int fl   = lane & 15;     // M row (A/D) / N col (B/D) sub-index
  const int hh   = lane >> 4;     // lane half

  // ---- stage W1 (256x128 f32 row-major) -> LDS transposed bf16 pairs ----
  {
    const int f   = tid;                  // feature column 0..127
    const int rt  = (f & 15) << 2;
    uint32_t* row = &sW[f * 128];
    for (int kp = 0; kp < 128; kp += 2) {
      float a0 = W1[(size_t)(2 * kp + 0) * HDIM + f];   // coalesced rows
      float a1 = W1[(size_t)(2 * kp + 1) * HDIM + f];
      float a2 = W1[(size_t)(2 * kp + 2) * HDIM + f];
      float a3 = W1[(size_t)(2 * kp + 3) * HDIM + f];
      uint2 pr;
      pr.x = pack2_bf16(a0, a1);
      pr.y = pack2_bf16(a2, a3);
      // kp even, rot multiple of 4 -> pair never splits across the &127 wrap
      *(uint2*)&row[(kp + rt) & 127] = pr;   // ds_store_b64
    }
  }
  __syncthreads();

  // per-lane slices of b1 / W2 across the 8 feature tiles
  float b1v[8], w2v[8];
#pragma unroll
  for (int n = 0; n < 8; ++n) {
    b1v[n] = b1[n * 16 + fl];
    w2v[n] = W2[n * 16 + fl];   // W2 is [128,1]
  }
  const float b2v = b2[0];
  const int rot = fl << 2;

  const int numTiles = (E + 127) >> 7;   // 128 edges per block iteration
  for (int tile = blockIdx.x; tile < numTiles; tile += gridDim.x) {
    const int e0  = tile * 128 + wv * 32;   // this wave: 32 edges = 2 M-tiles
    const int m0  = e0 + fl;
    const int m1  = e0 + 16 + fl;
    const int cm0 = m0 < E ? m0 : E - 1;
    const int cm1 = m1 < E ? m1 : E - 1;

    const int r0 = eidx[cm0], c0i = eidx[E + cm0];
    const int r1 = eidx[cm1], c1i = eidx[E + cm1];
    const float* u0 = zu + (size_t)r0  * HDIM;
    const float* v0 = zm + (size_t)c0i * HDIM;
    const float* u1 = zu + (size_t)r1  * HDIM;
    const float* v1 = zm + (size_t)c1i * HDIM;

    // persistent state: 8 feature-tile accumulators per M-tile (128 VGPRs)
    v8f acc0[8], acc1[8];
#pragma unroll
    for (int n = 0; n < 8; ++n) { acc0[n] = {}; acc1[n] = {}; }

#pragma unroll
    for (int k = 0; k < 8; ++k) {          // K = 256 in steps of 32
      FragBF a0, a1;                        // transient A fragments
      const float* p0 = (k < 4) ? (u0 + k * 32) : (v0 + (k - 4) * 32);
      const float* p1 = (k < 4) ? (u1 + k * 32) : (v1 + (k - 4) * 32);
      gather_frag(a0, p0, hh);
      gather_frag(a1, p1, hh);

      const int kp0 = k * 16 + hh * 4;     // this lane-half's first K-pair
#pragma unroll
      for (int n = 0; n < 8; ++n) {        // 8 feature tiles of 16
        const int f = n * 16 + fl;         // B: lane holds column f
        FragBF bfr;                         // B 32x16: VGPR v -> K = 32k+16*(v/4)+8h+2*(v%4)
        bfr.q[0] = *(const uint4*)&sW[f * 128 + ((kp0 + 0 + rot) & 127)];
        bfr.q[1] = *(const uint4*)&sW[f * 128 + ((kp0 + 8 + rot) & 127)];
        acc0[n] = __builtin_amdgcn_wmma_f32_16x16x32_bf16(
                    false, a0.bf, false, bfr.bf, (short)0, acc0[n], false, false);
        acc1[n] = __builtin_amdgcn_wmma_f32_16x16x32_bf16(
                    false, a1.bf, false, bfr.bf, (short)0, acc1[n], false, false);
      }
    }

    // fused epilogue: +b1, ReLU, * W2 slice, per-edge partial dot
    float pacc0[8], pacc1[8];
#pragma unroll
    for (int r = 0; r < 8; ++r) { pacc0[r] = 0.0f; pacc1[r] = 0.0f; }
#pragma unroll
    for (int n = 0; n < 8; ++n) {
#pragma unroll
      for (int r = 0; r < 8; ++r) {
        pacc0[r] += fmaxf(acc0[n][r] + b1v[n], 0.0f) * w2v[n];
        pacc1[r] += fmaxf(acc1[n][r] + b1v[n], 0.0f) * w2v[n];
      }
    }

    // reduce the 16 feature-lanes of each half (D: lane holds N=fl, M=8*hh+r)
#pragma unroll
    for (int r = 0; r < 8; ++r) {
      float s0 = pacc0[r], s1 = pacc1[r];
      s0 += __shfl_xor(s0, 1, 32); s0 += __shfl_xor(s0, 2, 32);
      s0 += __shfl_xor(s0, 4, 32); s0 += __shfl_xor(s0, 8, 32);
      s1 += __shfl_xor(s1, 1, 32); s1 += __shfl_xor(s1, 2, 32);
      s1 += __shfl_xor(s1, 4, 32); s1 += __shfl_xor(s1, 8, 32);
      pacc0[r] = s0; pacc1[r] = s1;
    }
    if (fl == 0) {
#pragma unroll
      for (int r = 0; r < 8; ++r) {
        int ma = e0 + hh * 8 + r;
        int mb = e0 + 16 + hh * 8 + r;
        if (ma < E) out[ma] = pacc0[r] + b2v;
        if (mb < E) out[mb] = pacc1[r] + b2v;
      }
    }
  }
}

extern "C" void kernel_launch(void* const* d_in, const int* in_sizes, int n_in,
                              void* d_out, int out_size, void* d_ws, size_t ws_size,
                              hipStream_t stream) {
  (void)n_in; (void)d_ws; (void)ws_size; (void)out_size;
  const float* zu   = (const float*)d_in[0];
  const float* zm   = (const float*)d_in[1];
  const int*   eidx = (const int*)d_in[2];
  const float* W1   = (const float*)d_in[3];
  const float* b1   = (const float*)d_in[4];
  const float* W2   = (const float*)d_in[5];
  const float* b2   = (const float*)d_in[6];
  float* out = (float*)d_out;

  const int E = in_sizes[2] / 2;
  int numTiles = (E + 127) / 128;
  // 1024 blocks: ~8 tiles each -> W1 staging L2 traffic amortized 8x,
  // still 4096 wave32 waves of parallelism.
  int blocks = numTiles < 1024 ? numTiles : 1024;
  if (blocks < 1) blocks = 1;

  edge_decoder_wmma<<<blocks, 128, 0, stream>>>(zu, zm, eidx, W1, b1, W2, b2, out, E);
}
